// BlockMoE_14233521619323
// MI455X (gfx1250) — compile-verified, matrix operand order checked by reading, not run on hardware
//
#include <hip/hip_runtime.h>
#include <hip/hip_bf16.h>

typedef __attribute__((ext_vector_type(16))) __bf16 v16bf;
typedef __attribute__((ext_vector_type(8)))  float  v8f;

union Frag {
    v16bf  v;
    __bf16 h[16];
    uint4  q[2];
};

#define B_ROWS 65536
#define FULL   1024
#define NEXP   64
#define WBLK   16
#define HID    64
#define GH     32
#define WAVES  4

// workspace byte offsets (bf16 fragment-packed weights)
#define WS_GW1 0u        // 64 frags  * 1024B
#define WS_GW2 65536u    // 4 frags
#define WS_EW1 69632u    // 256 frags
#define WS_EW2 331776u   // 256 frags
#define WS_EW3 593920u   // 64*32 bf16

__device__ __forceinline__ Frag load_frag(const unsigned char* base, int lane) {
    Frag f;
    const uint4* p = reinterpret_cast<const uint4*>(base) + (lane << 1);
    f.q[0] = p[0];
    f.q[1] = p[1];
    return f;
}

// ---------------------------------------------------------------------------
// Prep: convert fp32 weights -> bf16, packed into per-lane WMMA B-fragment
// layout. Fragment = 32x16 (KxN) tile, 1KB; lane l holds col n=(l&15)+nt*16,
// halfs j<8 -> K = kb+j, j>=8 -> K = kb+16+(j-8), kb = (l>>4)*8.
// ---------------------------------------------------------------------------
__global__ __launch_bounds__(32) void moe_prep(
    const float* __restrict__ gw1, const float* __restrict__ gw2,
    const float* __restrict__ ew1, const float* __restrict__ ew2,
    const float* __restrict__ ew3, unsigned char* __restrict__ ws) {
    const int bid  = blockIdx.x;
    const int lane = threadIdx.x;
    const int n16  = lane & 15;
    const int kb   = (lane >> 4) << 3;

    if (bid < 64) {                       // gw1 [1024,32]: 32 Ktiles x 2 Ntiles
        int kt = bid >> 1, nt = bid & 1;
        int n = n16 + nt * 16;
        __bf16* dst = (__bf16*)(ws + WS_GW1 + ((unsigned)bid << 10)) + lane * 16;
        #pragma unroll
        for (int j = 0; j < 16; ++j) {
            int k = kt * 32 + kb + ((j >> 3) << 4) + (j & 7);
            dst[j] = (__bf16)gw1[k * GH + n];
        }
    } else if (bid < 68) {                // gw2 [32,64]: 1 Ktile x 4 Ntiles
        int nt = bid - 64;
        int n = n16 + nt * 16;
        __bf16* dst = (__bf16*)(ws + WS_GW2 + ((unsigned)nt << 10)) + lane * 16;
        #pragma unroll
        for (int j = 0; j < 16; ++j) {
            int k = kb + ((j >> 3) << 4) + (j & 7);
            dst[j] = (__bf16)gw2[k * NEXP + n];
        }
    } else if (bid < 324) {               // ew1 [e][16,64]: K padded 16->32
        int f = bid - 68, e = f >> 2, nt = f & 3;
        int n = n16 + nt * 16;
        __bf16* dst = (__bf16*)(ws + WS_EW1 + ((unsigned)f << 10)) + lane * 16;
        #pragma unroll
        for (int j = 0; j < 16; ++j) {
            int k = kb + ((j >> 3) << 4) + (j & 7);
            float v = (k < WBLK) ? ew1[(e * WBLK + k) * HID + n] : 0.f;
            dst[j] = (__bf16)v;
        }
    } else if (bid < 580) {               // ew2 [e][64,32]: 2 Ktiles x 2 Ntiles
        int f = bid - 324, e = f >> 2, kt = (f >> 1) & 1, nt = f & 1;
        int n = n16 + nt * 16;
        __bf16* dst = (__bf16*)(ws + WS_EW2 + ((unsigned)f << 10)) + lane * 16;
        #pragma unroll
        for (int j = 0; j < 16; ++j) {
            int k = kt * 32 + kb + ((j >> 3) << 4) + (j & 7);
            dst[j] = (__bf16)ew2[(e * HID + k) * (HID / 2) + n];
        }
    } else {                              // ew3 [e][32] packed bf16
        __bf16* d3 = (__bf16*)(ws + WS_EW3);
        for (int e = 0; e < NEXP; ++e)
            d3[e * 32 + lane] = (__bf16)ew3[e * 32 + lane];
    }
}

// ---------------------------------------------------------------------------
// Main: one wave per 16-row tile. Stream x once (fp32->bf16 into LDS) fused
// with gate GEMM, then 64-expert block-MLP via WMMA + shuffle reductions.
// ---------------------------------------------------------------------------
__global__ __launch_bounds__(WAVES * 32) void moe_main(
    const float* __restrict__ x,
    const float* __restrict__ gb1, const float* __restrict__ gb2,
    const float* __restrict__ eb1, const float* __restrict__ eb2,
    const float* __restrict__ eb3,
    const unsigned char* __restrict__ ws,
    float* __restrict__ out) {

    __shared__ __attribute__((aligned(16))) __bf16 xs[WAVES][16][1032]; // padded: conflict-free b128
    __shared__ __attribute__((aligned(16))) __bf16 hb[WAVES][16][72];   // wave-private bounce buffer

    const int wave = threadIdx.x >> 5;
    const int lane = threadIdx.x & 31;
    const int row  = lane & 15;       // A-frag source row / D-layout column
    const int half = lane >> 4;       // D-layout row group (m = r + 8*half)
    const int kb   = half << 3;       // A-frag K sub-chunk base
    const int rowbase = blockIdx.x * (WAVES * 16) + wave * 16;

    const float* xrow = x + (size_t)(rowbase + row) * FULL;

    const v8f z = {0.f, 0.f, 0.f, 0.f, 0.f, 0.f, 0.f, 0.f};
    v8f g1[2] = {z, z};

    // ---- gate layer 1: [16,1024] @ [1024,32]; stage bf16 x into LDS ----
    for (int kt = 0; kt < 32; ++kt) {
        int c0 = kt * 32 + kb;
        float4 f0 = *(const float4*)(xrow + c0);
        float4 f1 = *(const float4*)(xrow + c0 + 4);
        float4 f2 = *(const float4*)(xrow + c0 + 16);
        float4 f3 = *(const float4*)(xrow + c0 + 20);
        if (kt < 30) __builtin_prefetch(xrow + c0 + 64, 0, 1);
        float t[16] = {f0.x, f0.y, f0.z, f0.w, f1.x, f1.y, f1.z, f1.w,
                       f2.x, f2.y, f2.z, f2.w, f3.x, f3.y, f3.z, f3.w};
        Frag a;
        #pragma unroll
        for (int j = 0; j < 16; ++j) a.h[j] = (__bf16)t[j];
        *(uint4*)&xs[wave][row][c0]      = a.q[0];
        *(uint4*)&xs[wave][row][c0 + 16] = a.q[1];

        Frag b0 = load_frag(ws + WS_GW1 + ((unsigned)(kt * 2 + 0) << 10), lane);
        Frag b1 = load_frag(ws + WS_GW1 + ((unsigned)(kt * 2 + 1) << 10), lane);
        g1[0] = __builtin_amdgcn_wmma_f32_16x16x32_bf16(false, a.v, false, b0.v, (short)0, g1[0], false, false);
        g1[1] = __builtin_amdgcn_wmma_f32_16x16x32_bf16(false, a.v, false, b1.v, (short)0, g1[1], false, false);
    }

    // bias + relu, bounce D-layout -> row-major LDS (cols 0..31)
    #pragma unroll
    for (int nt = 0; nt < 2; ++nt) {
        int col = (lane & 15) + nt * 16;
        float bb = gb1[col];
        #pragma unroll
        for (int r = 0; r < 8; ++r) {
            float vv = fmaxf(g1[nt][r] + bb, 0.f);
            hb[wave][r + 8 * half][col] = (__bf16)vv;
        }
    }

    // ---- gate layer 2: [16,32] @ [32,64] (LDS in-order: store->load safe) ----
    Frag ag;
    ag.q[0] = *(const uint4*)&hb[wave][row][kb];
    ag.q[1] = *(const uint4*)&hb[wave][row][kb + 16];

    v8f wsm[4];
    #pragma unroll
    for (int nt = 0; nt < 4; ++nt) {
        Frag b = load_frag(ws + WS_GW2 + ((unsigned)nt << 10), lane);
        wsm[nt] = __builtin_amdgcn_wmma_f32_16x16x32_bf16(false, ag.v, false, b.v, (short)0, z, false, false);
        float bb = gb2[(lane & 15) + nt * 16];
        #pragma unroll
        for (int r = 0; r < 8; ++r) wsm[nt][r] += bb;
    }

    // ---- softmax over 64 experts per row (16-lane butterfly per half) ----
    #pragma unroll
    for (int r = 0; r < 8; ++r) {
        float v0 = wsm[0][r], v1 = wsm[1][r], v2 = wsm[2][r], v3 = wsm[3][r];
        float mx = fmaxf(fmaxf(v0, v1), fmaxf(v2, v3));
        #pragma unroll
        for (int m = 1; m < 16; m <<= 1) mx = fmaxf(mx, __shfl_xor(mx, m, 32));
        float e0 = __expf(v0 - mx), e1 = __expf(v1 - mx);
        float e2 = __expf(v2 - mx), e3 = __expf(v3 - mx);
        float s = e0 + e1 + e2 + e3;
        #pragma unroll
        for (int m = 1; m < 16; m <<= 1) s += __shfl_xor(s, m, 32);
        float inv = 1.0f / s;
        wsm[0][r] = e0 * inv; wsm[1][r] = e1 * inv;
        wsm[2][r] = e2 * inv; wsm[3][r] = e3 * inv;
    }

    // ---- expert loop ----
    float outacc[8] = {0.f, 0.f, 0.f, 0.f, 0.f, 0.f, 0.f, 0.f};
    for (int e = 0; e < NEXP; ++e) {
        // broadcast w[m,e] to all lanes of its half
        int ef = e >> 4;
        int srcl = (e & 15) + (half << 4);
        float wv[8];
        #pragma unroll
        for (int r = 0; r < 8; ++r) {
            float t = (ef == 0) ? wsm[0][r] : (ef == 1) ? wsm[1][r]
                    : (ef == 2) ? wsm[2][r] : wsm[3][r];
            wv[r] = __shfl(t, srcl, 32);
        }

        // L1: A = x[:, e*16:(e+1)*16], K padded to 32 with zeros
        Frag a1;
        a1.q[0] = *(const uint4*)&xs[wave][row][e * WBLK + kb];
        a1.q[1] = make_uint4(0u, 0u, 0u, 0u);

        #pragma unroll
        for (int nt = 0; nt < 4; ++nt) {
            Frag b = load_frag(ws + WS_EW1 + ((unsigned)(e * 4 + nt) << 10), lane);
            v8f h = __builtin_amdgcn_wmma_f32_16x16x32_bf16(false, a1.v, false, b.v, (short)0, z, false, false);
            int col = (lane & 15) + nt * 16;
            float bb = eb1[e * HID + col];
            #pragma unroll
            for (int r = 0; r < 8; ++r) {
                float vv = fmaxf(h[r] + bb, 0.f);
                hb[wave][r + 8 * half][col] = (__bf16)vv;
            }
        }

        // L2: [16,64] @ [64,32], K = 2 tiles
        Frag a2k0, a2k1;
        a2k0.q[0] = *(const uint4*)&hb[wave][row][kb];
        a2k0.q[1] = *(const uint4*)&hb[wave][row][kb + 16];
        a2k1.q[0] = *(const uint4*)&hb[wave][row][32 + kb];
        a2k1.q[1] = *(const uint4*)&hb[wave][row][32 + kb + 16];

        v8f h2[2];
        #pragma unroll
        for (int nt = 0; nt < 2; ++nt) {
            Frag bk0 = load_frag(ws + WS_EW2 + ((unsigned)(e * 4 + nt) << 10), lane);
            Frag bk1 = load_frag(ws + WS_EW2 + ((unsigned)(e * 4 + 2 + nt) << 10), lane);
            v8f acc = __builtin_amdgcn_wmma_f32_16x16x32_bf16(false, a2k0.v, false, bk0.v, (short)0, z, false, false);
            acc = __builtin_amdgcn_wmma_f32_16x16x32_bf16(false, a2k1.v, false, bk1.v, (short)0, acc, false, false);
            int col = (lane & 15) + nt * 16;
            float bb = eb2[e * (HID / 2) + col];
            #pragma unroll
            for (int r = 0; r < 8; ++r) h2[nt][r] = fmaxf(acc[r] + bb, 0.f);
        }

        // L3: dot(32) via butterfly reduce; weighted accumulate
        const __bf16* w3 = (const __bf16*)(ws + WS_EW3) + e * 32;
        float c0 = (float)w3[lane & 15];
        float c1 = (float)w3[16 + (lane & 15)];
        float b3 = eb3[e];
        #pragma unroll
        for (int r = 0; r < 8; ++r) {
            float t = h2[0][r] * c0 + h2[1][r] * c1;
            #pragma unroll
            for (int m = 1; m < 16; m <<= 1) t += __shfl_xor(t, m, 32);
            outacc[r] += wv[r] * (t + b3);
        }
    }

    if ((lane & 15) == 0) {
        #pragma unroll
        for (int r = 0; r < 8; ++r)
            out[rowbase + 8 * half + r] = outacc[r];
    }
}

extern "C" void kernel_launch(void* const* d_in, const int* in_sizes, int n_in,
                              void* d_out, int out_size, void* d_ws, size_t ws_size,
                              hipStream_t stream) {
    const float* x   = (const float*)d_in[0];
    const float* gw1 = (const float*)d_in[1];
    const float* gb1 = (const float*)d_in[2];
    const float* gw2 = (const float*)d_in[3];
    const float* gb2 = (const float*)d_in[4];
    const float* ew1 = (const float*)d_in[5];
    const float* eb1 = (const float*)d_in[6];
    const float* ew2 = (const float*)d_in[7];
    const float* eb2 = (const float*)d_in[8];
    const float* ew3 = (const float*)d_in[9];
    const float* eb3 = (const float*)d_in[10];
    unsigned char* ws = (unsigned char*)d_ws;
    float* out = (float*)d_out;

    moe_prep<<<581, 32, 0, stream>>>(gw1, gw2, ew1, ew2, ew3, ws);
    moe_main<<<B_ROWS / (WAVES * 16), WAVES * 32, 0, stream>>>(
        x, gb1, gb2, eb1, eb2, eb3, ws, out);
}
